// RVQQuantizer_35562329211519
// MI455X (gfx1250) — matrix operand order repaired and checked by reading, not compiled, not run on hardware
//
#include <hip/hip_runtime.h>

#define B_DIM   16
#define C_DIM   512
#define T_DIM   2000
#define M_TOTAL (B_DIM * T_DIM)       // 32000 rows
#define K_CODES 1024
#define NUMQ    25
#define ZQ_ELEMS ((size_t)B_DIM * C_DIM * T_DIM)   // 16,384,000

#define ROWS_PER_BLOCK 64
#define WAVES_PER_BLOCK 4
#define LDA 520                        // bf16 elems per LDS row (1040B, 16B aligned)
#define LDS_BYTES (ROWS_PER_BLOCK * LDA * 2)

typedef __attribute__((ext_vector_type(8)))  __bf16 v8bf;
typedef __attribute__((ext_vector_type(16))) __bf16 v16bf;
typedef __attribute__((ext_vector_type(8)))  float  v8f;

__device__ __forceinline__ unsigned short f32_to_bf16_rne(float f) {
    unsigned u = __builtin_bit_cast(unsigned, f);
    unsigned r = (u + 0x7FFFu + ((u >> 16) & 1u)) >> 16;
    return (unsigned short)r;
}

__device__ __forceinline__ v16bf ld_b16x16(const __bf16* p) {
    v8bf lo = *(const v8bf*)(p);
    v8bf hi = *(const v8bf*)(p + 8);
    return __builtin_shufflevector(lo, hi,
        0, 1, 2, 3, 4, 5, 6, 7, 8, 9, 10, 11, 12, 13, 14, 15);
}

// ---------------------------------------------------------------------------
// Prep: z (B,C,T) -> residual (M=B*T, C) fp32
// ---------------------------------------------------------------------------
__global__ void rvq_transpose_z(const float* __restrict__ z,
                                float* __restrict__ residual) {
    size_t i = (size_t)blockIdx.x * 256 + threadIdx.x;
    if (i >= (size_t)M_TOTAL * C_DIM) return;
    int c = (int)(i & (C_DIM - 1));
    int m = (int)(i >> 9);
    int b = m / T_DIM;
    int t = m - b * T_DIM;
    residual[i] = z[((size_t)b * C_DIM + c) * T_DIM + t];
}

// ---------------------------------------------------------------------------
// Prep: codebooks fp32 -> bf16 (RNE)
// ---------------------------------------------------------------------------
__global__ void rvq_cb_to_bf16(const float* __restrict__ cb,
                               unsigned short* __restrict__ cb_bf) {
    size_t i = (size_t)blockIdx.x * 256 + threadIdx.x;
    if (i >= (size_t)NUMQ * K_CODES * C_DIM) return;
    cb_bf[i] = f32_to_bf16_rne(cb[i]);
}

// ---------------------------------------------------------------------------
// Prep: per-code squared norms ||c_k||^2
// ---------------------------------------------------------------------------
__global__ void rvq_cnorm(const float* __restrict__ cb,
                          float* __restrict__ cnorm) {
    int code = blockIdx.x * 256 + threadIdx.x;
    if (code >= NUMQ * K_CODES) return;
    const float* row = cb + (size_t)code * C_DIM;
    float s = 0.f;
    for (int j = 0; j < C_DIM; ++j) { float v = row[j]; s += v * v; }
    cnorm[code] = s;
}

// ---------------------------------------------------------------------------
// One RVQ stage: bf16 WMMA score GEMM + argmin + exact fp32 residual update.
// Block: 128 threads = 4 waves, 64 rows. Grid: 500 blocks.
// Each wave: 16 rows x 32 columns per outer iteration (two WMMA chains).
// ---------------------------------------------------------------------------
__global__ void rvq_stage(const unsigned short* __restrict__ cb_bf16_all,
                          const float* __restrict__ cb_f32_all,
                          const float* __restrict__ cnorm_all,
                          float* __restrict__ residual,
                          float* __restrict__ out_idx,
                          int q) {
    extern __shared__ __bf16 ldsA[];   // [ROWS_PER_BLOCK][LDA]

    const int tid  = threadIdx.x;
    const int lane = tid & 31;
    const int wave = tid >> 5;
    const int rowBase = blockIdx.x * ROWS_PER_BLOCK;

    const __bf16* cbq = reinterpret_cast<const __bf16*>(cb_bf16_all)
                        + (size_t)q * K_CODES * C_DIM;
    const float*  cbf = cb_f32_all + (size_t)q * K_CODES * C_DIM;
    const float*  cn  = cnorm_all + q * K_CODES;

    // Stage residual tile into LDS as bf16 (RNE).
    for (int i = tid; i < ROWS_PER_BLOCK * C_DIM; i += blockDim.x) {
        int r = i >> 9;
        int c = i & (C_DIM - 1);
        float f = residual[(size_t)(rowBase + r) * C_DIM + c];
        ldsA[r * LDA + c] = __builtin_bit_cast(__bf16, f32_to_bf16_rne(f));
    }
    __syncthreads();

    const int halfSel = (lane >> 4) & 1;   // 0: lanes 0-15, 1: lanes 16-31
    const int lrow    = lane & 15;
    const __bf16* aRow = &ldsA[(wave * 16 + lrow) * LDA];

    float bestv[8];
    int   besti[8];
#pragma unroll
    for (int r = 0; r < 8; ++r) { bestv[r] = 3.0e38f; besti[r] = 0; }

    for (int nt = 0; nt < K_CODES / 32; ++nt) {
        // Compiler memory barrier: force A fragments to be re-read from LDS
        // inside the loop (ds_load_b128) instead of being hoisted into
        // registers and spilled to scratch.
        asm volatile("" ::: "memory");

        const int ncol0 = nt * 32 + lrow;           // this lane's two columns
        const int ncol1 = ncol0 + 16;
        const __bf16* bCol0 = cbq + (size_t)ncol0 * C_DIM + halfSel * 16;
        const __bf16* bCol1 = cbq + (size_t)ncol1 * C_DIM + halfSel * 16;

        // Prefetch next column tile of the codebook (WGP-scope).
        if (nt + 1 < K_CODES / 32) {
            __builtin_prefetch(cbq + (size_t)(ncol0 + 32) * C_DIM, 0, 3);
            __builtin_prefetch(cbq + (size_t)(ncol1 + 32) * C_DIM, 0, 3);
        }

        v8f acc0 = {};
        v8f acc1 = {};
#pragma unroll
        for (int ct = 0; ct < C_DIM / 32; ++ct) {
            // A fragment: contiguous k order (matches B)
            //   lanes 0-15:  elems 0-15 = K 0-15
            //   lanes 16-31: elems 0-15 = K 16-31
            const int ka = ct * 32 + halfSel * 16;
            v8bf a0 = *(const v8bf*)(aRow + ka);
            v8bf a1 = *(const v8bf*)(aRow + ka + 8);
            v16bf a = __builtin_shufflevector(a0, a1,
                0, 1, 2, 3, 4, 5, 6, 7, 8, 9, 10, 11, 12, 13, 14, 15);

            // B fragment: contiguous k order (matches A) — lane holds column n=lane%16,
            // 16 contiguous K values (lanes 0-15: K 0-15, lanes 16-31: K 16-31).
            v16bf b0 = ld_b16x16(bCol0 + ct * 32);
            v16bf b1 = ld_b16x16(bCol1 + ct * 32);

            // One A load feeds two independent WMMA chains (ILP).
            acc0 = __builtin_amdgcn_wmma_f32_16x16x32_bf16(
                false, a, false, b0, (short)0, acc0, false, false);
            acc1 = __builtin_amdgcn_wmma_f32_16x16x32_bf16(
                false, a, false, b1, (short)0, acc1, false, false);
        }

        // d_k = ||c_k||^2 - 2 r.c_k   (||r||^2 constant per row -> dropped)
        const float cn0 = cn[ncol0];
        const float cn1 = cn[ncol1];
#pragma unroll
        for (int r = 0; r < 8; ++r) {
            float d0 = cn0 - 2.0f * acc0[r];
            if (d0 < bestv[r]) { bestv[r] = d0; besti[r] = ncol0; }
            float d1 = cn1 - 2.0f * acc1[r];
            if (d1 < bestv[r]) { bestv[r] = d1; besti[r] = ncol1; }
        }
    }

    // Cross-lane argmin within each 16-lane half (C/D layout:
    // lanes 0-15 hold rows r, lanes 16-31 hold rows r+8).
#pragma unroll
    for (int r = 0; r < 8; ++r) {
#pragma unroll
        for (int off = 1; off < 16; off <<= 1) {
            float ov = __shfl_xor(bestv[r], off, 32);
            int   oi = __shfl_xor(besti[r], off, 32);
            if (ov < bestv[r] || (ov == bestv[r] && oi < besti[r])) {
                bestv[r] = ov; besti[r] = oi;
            }
        }
    }

    // Exact fp32 residual update + index write (16 rows per wave).
#pragma unroll
    for (int lr = 0; lr < 16; ++lr) {
        const int src = (lr < 8) ? 0 : 16;
        const int idx = __shfl(besti[lr & 7], src, 32);
        const int grow = rowBase + wave * 16 + lr;
        if (lane == 0)
            out_idx[(size_t)q * M_TOTAL + grow] = (float)idx;
        const float* crow = cbf + (size_t)idx * C_DIM;
        float* rrow = residual + (size_t)grow * C_DIM;
        for (int ch = lane; ch < C_DIM; ch += 32)
            rrow[ch] -= crow[ch];
    }
}

// ---------------------------------------------------------------------------
// Finalize: z_q(b,c,t) = z(b,c,t) - residual_final(m=b*T+t, c)
// ---------------------------------------------------------------------------
__global__ void rvq_finalize(const float* __restrict__ z,
                             const float* __restrict__ residual,
                             float* __restrict__ zq) {
    size_t i = (size_t)blockIdx.x * 256 + threadIdx.x;
    if (i >= ZQ_ELEMS) return;
    int t = (int)(i % T_DIM);
    size_t tmp = i / T_DIM;
    int c = (int)(tmp % C_DIM);
    int b = (int)(tmp / C_DIM);
    size_t m = (size_t)b * T_DIM + t;
    zq[i] = z[i] - residual[m * C_DIM + c];
}

// ---------------------------------------------------------------------------
extern "C" void kernel_launch(void* const* d_in, const int* in_sizes, int n_in,
                              void* d_out, int out_size, void* d_ws, size_t ws_size,
                              hipStream_t stream) {
    const float* z  = (const float*)d_in[0];   // (B, C, T)
    const float* cb = (const float*)d_in[1];   // (NUMQ, K, C)
    float* out = (float*)d_out;                // [z_q (B*C*T) | indices (NUMQ*B*T)]

    char* ws = (char*)d_ws;
    float*          residual = (float*)ws;                      // 65,536,000 B
    unsigned short* cb_bf    = (unsigned short*)(ws + 65536000);// 26,214,400 B
    float*          cnorm    = (float*)(ws + 65536000 + 26214400);

    // Allow >64KB dynamic LDS for the stage kernel (66,560 B).
    (void)hipFuncSetAttribute((const void*)rvq_stage,
                              hipFuncAttributeMaxDynamicSharedMemorySize,
                              LDS_BYTES);

    // Prep
    rvq_transpose_z<<<(int)((ZQ_ELEMS + 255) / 256), 256, 0, stream>>>(z, residual);
    rvq_cb_to_bf16<<<(NUMQ * K_CODES * C_DIM + 255) / 256, 256, 0, stream>>>(cb, cb_bf);
    rvq_cnorm<<<(NUMQ * K_CODES + 255) / 256, 256, 0, stream>>>(cb, cnorm);

    // 25 sequential RVQ stages (residual dependency -> separate launches)
    float* out_idx = out + ZQ_ELEMS;
    for (int q = 0; q < NUMQ; ++q) {
        rvq_stage<<<M_TOTAL / ROWS_PER_BLOCK, WAVES_PER_BLOCK * 32, LDS_BYTES, stream>>>(
            cb_bf, cb, cnorm, residual, out_idx, q);
    }

    // z_q = z - final residual
    rvq_finalize<<<(int)((ZQ_ELEMS + 255) / 256), 256, 0, stream>>>(z, residual, out);
}
